// KoopmanAE_54099408060868
// MI455X (gfx1250) — matrix operand order repaired and checked by reading, not compiled, not run on hardware
//
#include <hip/hip_runtime.h>

typedef __attribute__((ext_vector_type(16))) _Float16 v16h;
typedef __attribute__((ext_vector_type(8)))  _Float16 v8h;
typedef __attribute__((ext_vector_type(2)))  __fp16   v2fp16;  // return type of cvt_pkrtz
typedef __attribute__((ext_vector_type(8)))  float    v8f;

#define B_DIM 8192
#define D_IN  4096
#define HID   4096
#define F_LAT 1024   // latent = 2*F_LAT = 2048

static __device__ __forceinline__ unsigned pack2(float a, float b) {
  v2fp16 h = __builtin_amdgcn_cvt_pkrtz(a, b);   // v_cvt_pk_rtz_f16_f32
  unsigned u;
  __builtin_memcpy(&u, &h, 4);
  return u;
}

// Branch-free tanh: avoids libm's divergent saveexec path in the epilogue.
// tanh(x) = sign(x) * (1 - e) / (1 + e), e = exp(-2|x|) in (0, 1] -> no overflow.
static __device__ __forceinline__ float fast_tanh(float x) {
  const float ax = __builtin_fabsf(x);
  const float e  = __expf(-2.0f * ax);
  const float t  = __fdividef(1.0f - e, 1.0f + e);
  return __builtin_copysignf(t, x);
}

// C[M,N] = op(A[M,K] @ Bw[K,N] + bias), A row-major (f32 or f16), Bw row-major f32.
// Tiles: 128x128x32, 256 threads (8 waves), wave grid 4(M) x 2(N),
// each wave computes 2x4 = 8 v_wmma_f32_16x16x32_f16 tiles per k-step.
// Software pipeline: global loads for tile k+1 issued before compute of tile k.
template<bool A16, bool DO_TANH, bool OUT32, bool OUT16>
__global__ __launch_bounds__(256) void gemm_wmma(
    const void* __restrict__ Aptr, const float* __restrict__ Bw,
    const float* __restrict__ bias, float* __restrict__ out32,
    _Float16* __restrict__ out16, int M, int N, int K)
{
  constexpr int BM = 128, BN = 128, BK = 32;
  constexpr int ASTR = 40;  // halfs per A row: 32 + 8 pad (80B stride: 16B aligned, conflict-free b128)
  constexpr int BSTR = 20;  // half2 per B col: 16 + 4 pad (80B stride); layout [col][khalf][k2in8]
  __shared__ _Float16 As[BM * ASTR];
  __shared__ unsigned Bs[BN * BSTR];

  const int tid  = threadIdx.x;
  const int lane = tid & 31;
  const int wv   = tid >> 5;
  const int half = lane >> 4;   // 0: lanes 0-15, 1: lanes 16-31
  const int l16  = lane & 15;
  const int wm   = wv >> 1;     // 0..3 (M direction)
  const int wn   = wv & 1;      // 0..1 (N direction)
  const int mBase = blockIdx.y * BM;
  const int nBase = blockIdx.x * BN;

  const float*    Ap32 = (const float*)Aptr;
  const _Float16* Ap16 = (const _Float16*)Aptr;

  // ---- staging registers (tile in flight) ----
  float4 sa32[4];
  v8h    sa16[2];
  float4 sb0[2], sb1[2];

  auto global_load = [&](int k0) {
    if constexpr (!A16) {
      #pragma unroll
      for (int i = 0; i < 4; ++i) {
        int idx = tid + i * 256;                 // 0..1023 float4-chunks
        int r = idx >> 3;
        int c = (idx & 7) << 2;
        sa32[i] = *(const float4*)(Ap32 + (size_t)(mBase + r) * K + k0 + c);
      }
    } else {
      #pragma unroll
      for (int i = 0; i < 2; ++i) {
        int idx = tid + i * 256;                 // 0..511 16B-chunks
        int r = idx >> 2;
        int c = (idx & 3) << 3;
        sa16[i] = *(const v8h*)(Ap16 + (size_t)(mBase + r) * K + k0 + c);
      }
    }
    #pragma unroll
    for (int i = 0; i < 2; ++i) {
      int t2 = tid + i * 256;                    // 0..511 quad-tasks
      int k2 = t2 >> 5;                          // 0..15
      int c  = (t2 & 31) << 2;                   // 0..124
      const float* p = Bw + (size_t)(k0 + 2 * k2) * N + nBase + c;
      sb0[i] = *(const float4*)p;
      sb1[i] = *(const float4*)(p + N);
    }
  };

  auto lds_store = [&]() {
    if constexpr (!A16) {
      #pragma unroll
      for (int i = 0; i < 4; ++i) {
        int idx = tid + i * 256;
        int r = idx >> 3;
        int c = (idx & 7) << 2;
        unsigned* dst = (unsigned*)&As[r * ASTR + c];
        dst[0] = pack2(sa32[i].x, sa32[i].y);
        dst[1] = pack2(sa32[i].z, sa32[i].w);
      }
    } else {
      #pragma unroll
      for (int i = 0; i < 2; ++i) {
        int idx = tid + i * 256;
        int r = idx >> 2;
        int c = (idx & 3) << 3;
        *(v8h*)&As[r * ASTR + c] = sa16[i];
      }
    }
    #pragma unroll
    for (int i = 0; i < 2; ++i) {
      int t2 = tid + i * 256;
      int k2 = t2 >> 5;
      int c  = (t2 & 31) << 2;
      const int khalf = k2 >> 3;                 // K 0..15 vs 16..31
      const int k8    = k2 & 7;                  // fragment half2 index
      // Bs[col][khalf][k8] : fragment-order so reads are two contiguous b128
      Bs[(c + 0) * BSTR + khalf * 8 + k8] = pack2(sb0[i].x, sb1[i].x);
      Bs[(c + 1) * BSTR + khalf * 8 + k8] = pack2(sb0[i].y, sb1[i].y);
      Bs[(c + 2) * BSTR + khalf * 8 + k8] = pack2(sb0[i].z, sb1[i].z);
      Bs[(c + 3) * BSTR + khalf * 8 + k8] = pack2(sb0[i].w, sb1[i].w);
    }
  };

  v8f acc[2][4] = {};

  global_load(0);
  int k0 = 0;
  for (;;) {
    lds_store();
    __syncthreads();                             // LDS tile visible

    const int kn = k0 + BK;
    const bool more = kn < K;
    if (more) {
      global_load(kn);                           // overlap next tile's loads with compute
      if (kn + BK < K) {                         // prefetch tile k+2 into L2
        __builtin_prefetch(Bw + (size_t)(kn + BK) * N + nBase + (tid & 127), 0, 2);
        if constexpr (!A16)
          __builtin_prefetch(Ap32 + (size_t)(mBase + (tid >> 1)) * K + kn + BK, 0, 2);
        else
          __builtin_prefetch(Ap16 + (size_t)(mBase + (tid >> 1)) * K + kn + BK, 0, 2);
      }
    }

    // ---- fragments (contiguous k order: k = 16*half + s, same for A and B) ----
    // A 16x32: lane(l16) = row M; half=0 -> K 0..15, half=1 -> K 16..31
    v16h afrag[2];
    #pragma unroll
    for (int mi = 0; mi < 2; ++mi) {
      const _Float16* ar = &As[(wm * 32 + mi * 16 + l16) * ASTR];
      v8h lo = *(const v8h*)(ar + 16 * half);
      v8h hi = *(const v8h*)(ar + 16 * half + 8);
      afrag[mi] = __builtin_shufflevector(lo, hi,
          0, 1, 2, 3, 4, 5, 6, 7, 8, 9, 10, 11, 12, 13, 14, 15);
    }
    // B 32x16: VGPR n holds (K=2n, 2n+1) + 16*half, N = l16 -> two contiguous b128 per lane
    v16h bfrag[4];
    #pragma unroll
    for (int ni = 0; ni < 4; ++ni) {
      const unsigned* br = &Bs[(wn * 64 + ni * 16 + l16) * BSTR + half * 8];
      v8h lo = *(const v8h*)(br + 0);
      v8h hi = *(const v8h*)(br + 4);
      bfrag[ni] = __builtin_shufflevector(lo, hi,
          0, 1, 2, 3, 4, 5, 6, 7, 8, 9, 10, 11, 12, 13, 14, 15);
    }

    #pragma unroll
    for (int mi = 0; mi < 2; ++mi)
      #pragma unroll
      for (int ni = 0; ni < 4; ++ni)
        acc[mi][ni] = __builtin_amdgcn_wmma_f32_16x16x32_f16(
            false, afrag[mi], false, bfrag[ni], (short)0, acc[mi][ni], false, false);

    if (!more) break;
    k0 = kn;
    __syncthreads();                             // compute done before LDS overwrite
  }

  // ---- epilogue: bias (+tanh), store f32 and/or f16 ----
  // C/D layout: VGPR v, half h -> M = v + 8*h, N = l16
  #pragma unroll
  for (int mi = 0; mi < 2; ++mi) {
    const int rowB = mBase + wm * 32 + mi * 16 + 8 * half;
    #pragma unroll
    for (int ni = 0; ni < 4; ++ni) {
      const int col = nBase + wn * 64 + ni * 16 + l16;
      const float bv = bias[col];
      #pragma unroll
      for (int v = 0; v < 8; ++v) {
        float r = acc[mi][ni][v] + bv;
        if constexpr (DO_TANH) r = fast_tanh(r);
        const size_t o = (size_t)(rowB + v) * N + col;
        if constexpr (OUT32) out32[o] = r;
        if constexpr (OUT16) out16[o] = (_Float16)r;
      }
    }
  }
}

// Per-sample Koopman block rotation; one thread per (b, f) 2-vector.
__global__ __launch_bounds__(256) void koopman_rotate(
    const float* __restrict__ enc_x, const float* __restrict__ dt,
    const float* __restrict__ amp, const float* __restrict__ freq,
    float* __restrict__ enc_pred, _Float16* __restrict__ ep16)
{
  const int i = blockIdx.x * blockDim.x + threadIdx.x;
  if (i >= B_DIM * F_LAT) return;
  const int b = i / F_LAT;
  const int f = i - b * F_LAT;
  const float t = dt[b];
  const float a = powf(amp[f], t);
  const float ang = freq[f] * t;
  float s, c;
  sincosf(ang, &s, &c);
  c *= a;
  s *= a;
  const size_t o = (size_t)b * (2 * F_LAT) + 2 * f;
  const float2 eo = *(const float2*)(enc_x + o);
  float2 r;
  r.x = c * eo.x - s * eo.y;
  r.y = s * eo.x + c * eo.y;
  *(float2*)(enc_pred + o) = r;
  *(unsigned*)(ep16 + o) = pack2(r.x, r.y);
}

extern "C" void kernel_launch(void* const* d_in, const int* in_sizes, int n_in,
                              void* d_out, int out_size, void* d_ws, size_t ws_size,
                              hipStream_t stream) {
  const float* x      = (const float*)d_in[0];
  const float* dt     = (const float*)d_in[1];
  const float* W_enc1 = (const float*)d_in[2];
  const float* b_enc1 = (const float*)d_in[3];
  const float* W_enc2 = (const float*)d_in[4];
  const float* b_enc2 = (const float*)d_in[5];
  const float* amp    = (const float*)d_in[6];
  const float* freq   = (const float*)d_in[7];
  const float* W_dec1 = (const float*)d_in[8];
  const float* b_dec1 = (const float*)d_in[9];
  const float* W_dec2 = (const float*)d_in[10];
  const float* b_dec2 = (const float*)d_in[11];

  float* out       = (float*)d_out;
  float* full_pred = out;                                        // [B, D_IN]
  float* enc_pred  = out + (size_t)B_DIM * D_IN;                 // [B, 2F]
  float* enc_x     = enc_pred + (size_t)B_DIM * 2 * F_LAT;       // [B, 2F]

  _Float16* h16  = (_Float16*)d_ws;                              // [B, HID] f16 (h1 then h2)
  _Float16* ep16 = h16 + (size_t)B_DIM * HID;                    // [B, 2F] f16

  dim3 blk(256);
  // 1) h1 = tanh(x @ W_enc1 + b_enc1) -> f16 ws
  gemm_wmma<false, true, false, true>
      <<<dim3(HID / 128, B_DIM / 128), blk, 0, stream>>>(
          x, W_enc1, b_enc1, nullptr, h16, B_DIM, HID, D_IN);
  // 2) enc_x = h1 @ W_enc2 + b_enc2 -> f32 d_out
  gemm_wmma<true, false, true, false>
      <<<dim3((2 * F_LAT) / 128, B_DIM / 128), blk, 0, stream>>>(
          h16, W_enc2, b_enc2, enc_x, nullptr, B_DIM, 2 * F_LAT, HID);
  // 3) Koopman rotation -> enc_pred (f32 d_out) + f16 ws
  koopman_rotate<<<dim3((B_DIM * F_LAT) / 256), blk, 0, stream>>>(
      enc_x, dt, amp, freq, enc_pred, ep16);
  // 4) h2 = tanh(enc_pred @ W_dec1 + b_dec1) -> f16 ws
  gemm_wmma<true, true, false, true>
      <<<dim3(HID / 128, B_DIM / 128), blk, 0, stream>>>(
          ep16, W_dec1, b_dec1, nullptr, h16, B_DIM, HID, 2 * F_LAT);
  // 5) full_pred = h2 @ W_dec2 + b_dec2 -> f32 d_out
  gemm_wmma<true, false, true, false>
      <<<dim3(D_IN / 128, B_DIM / 128), blk, 0, stream>>>(
          h16, W_dec2, b_dec2, full_pred, nullptr, B_DIM, D_IN, HID);

  (void)in_sizes; (void)n_in; (void)out_size; (void)ws_size;
}
